// MaskedMultiheadAttention_7791070675439
// MI455X (gfx1250) — compile-verified, matrix operand order checked by reading, not run on hardware
//
#include <hip/hip_runtime.h>
#include <stdint.h>

#define T_SEQ 2048
#define BATCH 4
#define EMB   1024
#define NH    16
#define DH    64
#define ATTN_SCALE 0.125f   // 1/sqrt(64)

typedef __bf16 bf16;
typedef __attribute__((ext_vector_type(16))) __bf16 v16bf;
typedef __attribute__((ext_vector_type(8)))  __bf16 v8bf;
typedef __attribute__((ext_vector_type(4)))  __bf16 v4bf;
typedef __attribute__((ext_vector_type(8)))  float  v8f;
typedef __attribute__((ext_vector_type(4)))  float  f32x4;
typedef __attribute__((ext_vector_type(4)))  unsigned int u32x4;
typedef __attribute__((ext_vector_type(4)))  int i32x4;

union FragU { v16bf v; u32x4 h[2]; };

// ---- CDNA5 async global->LDS staging (guarded; falls back to load+store) ----
#if defined(__has_builtin)
#if __has_builtin(__builtin_amdgcn_global_load_async_to_lds_b128)
#define HAVE_ASYNC_LDS 1
#endif
#endif
#ifndef HAVE_ASYNC_LDS
#define HAVE_ASYNC_LDS 0
#endif

__device__ __forceinline__ void cp16_g2l(const bf16* g, bf16* l) {
#if HAVE_ASYNC_LDS
  __builtin_amdgcn_global_load_async_to_lds_b128((i32x4*)g, (i32x4*)l, 0, 0);
#else
  *(u32x4*)l = *(const u32x4*)g;
#endif
}
__device__ __forceinline__ void async_wait0() {
#if HAVE_ASYNC_LDS
  asm volatile("s_wait_asynccnt 0x0" ::: "memory");
#endif
}

// round-to-nearest-even f32 -> bf16 (bit-level)
__device__ __forceinline__ bf16 f2bf(float f) {
  unsigned u = __builtin_bit_cast(unsigned, f);
  unsigned r = (u + 0x7FFFu + ((u >> 16) & 1u)) >> 16;
  unsigned short s = (unsigned short)r;
  return __builtin_bit_cast(bf16, s);
}

__device__ __forceinline__ v8f wmma_bf16(v16bf a, v16bf b, v8f c) {
  return __builtin_amdgcn_wmma_f32_16x16x32_bf16(false, a, false, b, (short)0, c, false, false);
}

// 16x32 bf16 operand fragment from a row-major LDS tile.
// lane&15 = row, half = K-subset; two contiguous 8-bf16 runs per lane.
__device__ __forceinline__ v16bf load_frag(const bf16* base, int strideBf,
                                           int rowOff, int colOff) {
  const int lane = threadIdx.x & 31;
  const int half = lane >> 4;
  const bf16* p = base + (size_t)((lane & 15) + rowOff) * strideBf + colOff + half * 8;
  FragU f;
  f.h[0] = *(const u32x4*)(p);
  f.h[1] = *(const u32x4*)(p + 16);
  return f.v;
}

// ---------------------------------------------------------------------------
// Kernel 0: one-shot f32 -> bf16 conversion (bandwidth-trivial on 23.3 TB/s)
// ---------------------------------------------------------------------------
__global__ __launch_bounds__(256) void cvt_f32_bf16_kernel(
    const float* __restrict__ in, bf16* __restrict__ out, int n)
{
  int i = (blockIdx.x * 256 + threadIdx.x) * 8;
  if (i + 8 <= n) {
    f32x4 a = *(const f32x4*)(in + i);
    f32x4 b = *(const f32x4*)(in + i + 4);
    v8bf o;
    o[0] = f2bf(a.x); o[1] = f2bf(a.y); o[2] = f2bf(a.z); o[3] = f2bf(a.w);
    o[4] = f2bf(b.x); o[5] = f2bf(b.y); o[6] = f2bf(b.z); o[7] = f2bf(b.w);
    *(v8bf*)(out + i) = o;
  }
}

// ---------------------------------------------------------------------------
// Kernel 1: QKV projection, all-bf16, double-buffered async LDS staging.
// grid = (M/128, N/128, 3), block = 256 (8 waves, 4x2, 32x64 per wave)
// ---------------------------------------------------------------------------
__global__ __launch_bounds__(256) void qkv_proj_kernel(
    const bf16* __restrict__ xq, const bf16* __restrict__ xk,
    const bf16* __restrict__ xv, const bf16* __restrict__ wqkv,
    const float* __restrict__ b_qkv,
    bf16* __restrict__ q_out, bf16* __restrict__ k_out, bf16* __restrict__ v_out)
{
  __shared__ bf16 sA[2][128 * 40];
  __shared__ bf16 sB[2][128 * 40];
  const int tid  = threadIdx.x;
  const int lane = tid & 31, wave = tid >> 5;
  const int half = lane >> 4, l16 = lane & 15;
  const int which = blockIdx.z;
  const bf16* x    = (which == 0) ? xq : ((which == 1) ? xk : xv);
  const bf16* w    = wqkv + (size_t)which * EMB * EMB;
  const float* bias = b_qkv + which * EMB;
  bf16* outp        = (which == 0) ? q_out : ((which == 1) ? k_out : v_out);
  const int m0 = blockIdx.x * 128;
  const int n0 = blockIdx.y * 128;
  const int wm = wave & 3, wn = wave >> 2;

  auto stage = [&](int kk, bf16* dA, bf16* dB) {
#pragma unroll
    for (int i = 0; i < 2; ++i) {
      int idx = tid + i * 256;            // 512 chunks per tile
      int row = idx >> 2;
      int c   = (idx & 3) * 8;
      cp16_g2l(x + (size_t)(m0 + row) * EMB + kk + c, dA + row * 40 + c);
      cp16_g2l(w + (size_t)(n0 + row) * EMB + kk + c, dB + row * 40 + c);
    }
  };

  bf16 *curA = sA[0], *curB = sB[0], *nxtA = sA[1], *nxtB = sB[1];
  stage(0, curA, curB);
  v8f acc[2][4] = {};

#pragma unroll 1            // keep accumulators pinned; no unroll-induced copies
  for (int kt = 0; kt < EMB / 32; ++kt) {
    async_wait0();
    __syncthreads();
    if (kt + 1 < EMB / 32) stage((kt + 1) * 32, nxtA, nxtB);
    v16bf a0 = load_frag(curA, 40, wm * 32,      0);
    v16bf a1 = load_frag(curA, 40, wm * 32 + 16, 0);
#pragma unroll
    for (int j = 0; j < 4; ++j) {
      v16bf b = load_frag(curB, 40, wn * 64 + j * 16, 0);
      acc[0][j] = wmma_bf16(a0, b, acc[0][j]);
      acc[1][j] = wmma_bf16(a1, b, acc[1][j]);
    }
    bf16* t;
    t = curA; curA = nxtA; nxtA = t;
    t = curB; curB = nxtB; nxtB = t;
  }

  // epilogue: +bias, convert, scatter into [B,H,T,DH]
#pragma unroll
  for (int j = 0; j < 4; ++j) {
    int n = n0 + wn * 64 + j * 16 + l16;
    float bj = bias[n];
    int hh = n >> 6, dd = n & 63;
#pragma unroll
    for (int i = 0; i < 2; ++i) {
#pragma unroll
      for (int r = 0; r < 8; ++r) {
        int m = m0 + wm * 32 + i * 16 + r + half * 8;   // row in [T*B]
        int t = m >> 2, bb = m & 3;                      // x is [T,B,E]
        float val = acc[i][j][r] + bj;
        outp[(((size_t)bb * NH + hh) * T_SEQ + t) * DH + dd] = f2bf(val);
      }
    }
  }
}

// ---------------------------------------------------------------------------
// Kernel 2: streaming attention (flash-style, multiplicative mask).
// grid = (T/128, B*H), block = 256.  Each wave owns 16 query rows.
// ---------------------------------------------------------------------------
__global__ __launch_bounds__(256) void attention_kernel(
    const bf16* __restrict__ Q, const bf16* __restrict__ K,
    const bf16* __restrict__ V, const float* __restrict__ mask,
    bf16* __restrict__ ctx)
{
  __shared__ bf16 sK[32 * 72];     // 32 keys x 64 d (+8 pad)
  __shared__ bf16 sVt[64 * 40];    // V transposed: 64 d x 32 keys (+8 pad)
  __shared__ bf16 sP[8][16 * 40];  // per-wave P scratch: 16 q x 32 k (+8 pad)

  const int tid  = threadIdx.x;
  const int lane = tid & 31, wave = tid >> 5;
  const int half = lane >> 4, l16 = lane & 15;
  const int bh = blockIdx.y;
  const int b  = bh >> 4, h = bh & 15;
  const size_t base = (size_t)bh * T_SEQ * DH;
  const int q0 = blockIdx.x * 128;
  const int qw = q0 + wave * 16;

  // Persistent Q fragments for d in [0,32) and [32,64)
  v16bf aq[2];
  {
    const bf16* qp = Q + base + (size_t)(qw + l16) * DH;
#pragma unroll
    for (int s = 0; s < 2; ++s) {
      FragU f;
      f.h[0] = *(const u32x4*)(qp + s * 32 + half * 8);
      f.h[1] = *(const u32x4*)(qp + s * 32 + half * 8 + 16);
      aq[s] = f.v;
    }
  }

  float m_run[8], l_run[8];
  v8f o[4] = {};
#pragma unroll
  for (int r = 0; r < 8; ++r) { m_run[r] = -3.0e38f; l_run[r] = 0.0f; }

#pragma unroll 1
  for (int kc = 0; kc < T_SEQ; kc += 32) {
    __syncthreads();   // previous iteration's readers of sK/sVt are done
    // K: async 16B chunks (256 chunks, 1/thread); V: transposed scalar stores
    cp16_g2l(K + base + (size_t)(kc + (tid >> 3)) * DH + (tid & 7) * 8,
             sK + (tid >> 3) * 72 + (tid & 7) * 8);
#pragma unroll
    for (int i = 0; i < 2; ++i) {
      int idx = tid + i * 256;
      int row = idx >> 4;
      int c   = (idx & 15) * 4;
      v4bf vv = *(const v4bf*)(V + base + (size_t)(kc + row) * DH + c);
      sVt[(c + 0) * 40 + row] = vv.x;
      sVt[(c + 1) * 40 + row] = vv.y;
      sVt[(c + 2) * 40 + row] = vv.z;
      sVt[(c + 3) * 40 + row] = vv.w;
    }
    // prefetch next mask block (global_prefetch_b8; OOB wraps harmlessly)
    __builtin_prefetch(mask + (size_t)(qw + l16) * T_SEQ +
                           ((kc + 32) & (T_SEQ - 1)) + half * 16, 0, 1);
    async_wait0();
    __syncthreads();

    // S(16x32) = Q(16x64) K^T
    v8f sacc[2] = {};
#pragma unroll
    for (int j = 0; j < 2; ++j) {
      v16bf bk0 = load_frag(sK, 72, j * 16, 0);
      v16bf bk1 = load_frag(sK, 72, j * 16, 32);
      sacc[j] = wmma_bf16(aq[0], bk0, sacc[j]);
      sacc[j] = wmma_bf16(aq[1], bk1, sacc[j]);
    }

    // scale * mask, online softmax (rows live in 16-lane half-groups)
    float p0[8], p1[8], alpha[8];
    const int tqb = qw + half * 8;
#pragma unroll
    for (int r = 0; r < 8; ++r) {
      float s0 = sacc[0][r] * ATTN_SCALE * mask[(size_t)(tqb + r) * T_SEQ + kc + l16];
      float s1 = sacc[1][r] * ATTN_SCALE * mask[(size_t)(tqb + r) * T_SEQ + kc + 16 + l16];
      float mx = fmaxf(s0, s1);
#pragma unroll
      for (int d = 8; d >= 1; d >>= 1) mx = fmaxf(mx, __shfl_xor(mx, d, 32));
      float mnew = fmaxf(m_run[r], mx);
      float a    = __expf(m_run[r] - mnew);
      float e0 = __expf(s0 - mnew), e1 = __expf(s1 - mnew);
      float rs = e0 + e1;
#pragma unroll
      for (int d = 8; d >= 1; d >>= 1) rs += __shfl_xor(rs, d, 32);
      l_run[r] = l_run[r] * a + rs;
      m_run[r] = mnew;
      alpha[r] = a;
      p0[r] = e0; p1[r] = e1;
    }
#pragma unroll
    for (int f = 0; f < 4; ++f)
#pragma unroll
      for (int r = 0; r < 8; ++r) o[f][r] *= alpha[r];

    // C-layout -> A-layout for P via wave-private LDS bounce
    bf16* sPw = sP[wave];
#pragma unroll
    for (int r = 0; r < 8; ++r) {
      sPw[(r + half * 8) * 40 + l16]      = f2bf(p0[r]);
      sPw[(r + half * 8) * 40 + 16 + l16] = f2bf(p1[r]);
    }
    asm volatile("s_wait_dscnt 0x0" ::: "memory");  // wave-internal LDS RAW
    v16bf ap = load_frag(sPw, 40, 0, 0);
#pragma unroll
    for (int f = 0; f < 4; ++f) {
      v16bf bv = load_frag(sVt, 40, f * 16, 0);     // B: 32 keys x 16 d
      o[f] = wmma_bf16(ap, bv, o[f]);
    }
  }

  // normalize and store ctx bf16 at [B,T,E]
#pragma unroll
  for (int r = 0; r < 8; ++r) l_run[r] = 1.0f / l_run[r];
#pragma unroll
  for (int f = 0; f < 4; ++f) {
#pragma unroll
    for (int r = 0; r < 8; ++r) {
      int t = qw + r + half * 8;
      int d = f * 16 + l16;
      ctx[((size_t)b * T_SEQ + t) * EMB + h * DH + d] = f2bf(o[f][r] * l_run[r]);
    }
  }
}

// ---------------------------------------------------------------------------
// Kernel 3: output projection, all-bf16 operands, double-buffered staging.
// out[T,B,E](f32) = ctx[B,T,E](bf16) @ w_out^T + b_out
// ---------------------------------------------------------------------------
__global__ __launch_bounds__(256) void out_proj_kernel(
    const bf16* __restrict__ ctx, const bf16* __restrict__ wout,
    const float* __restrict__ b_out, float* __restrict__ out)
{
  __shared__ bf16 sA[2][128 * 40];
  __shared__ bf16 sB[2][128 * 40];
  const int tid  = threadIdx.x;
  const int lane = tid & 31, wave = tid >> 5;
  const int half = lane >> 4, l16 = lane & 15;
  const int m0 = blockIdx.x * 128;
  const int n0 = blockIdx.y * 128;
  const int wm = wave & 3, wn = wave >> 2;

  auto stage = [&](int kk, bf16* dA, bf16* dB) {
#pragma unroll
    for (int i = 0; i < 2; ++i) {
      int idx = tid + i * 256;
      int row = idx >> 2;
      int c   = (idx & 3) * 8;
      cp16_g2l(ctx  + (size_t)(m0 + row) * EMB + kk + c, dA + row * 40 + c);
      cp16_g2l(wout + (size_t)(n0 + row) * EMB + kk + c, dB + row * 40 + c);
    }
  };

  bf16 *curA = sA[0], *curB = sB[0], *nxtA = sA[1], *nxtB = sB[1];
  stage(0, curA, curB);
  v8f acc[2][4] = {};

#pragma unroll 1
  for (int kt = 0; kt < EMB / 32; ++kt) {
    async_wait0();
    __syncthreads();
    if (kt + 1 < EMB / 32) stage((kt + 1) * 32, nxtA, nxtB);
    v16bf a0 = load_frag(curA, 40, wm * 32,      0);
    v16bf a1 = load_frag(curA, 40, wm * 32 + 16, 0);
#pragma unroll
    for (int j = 0; j < 4; ++j) {
      v16bf b = load_frag(curB, 40, wn * 64 + j * 16, 0);
      acc[0][j] = wmma_bf16(a0, b, acc[0][j]);
      acc[1][j] = wmma_bf16(a1, b, acc[1][j]);
    }
    bf16* t;
    t = curA; curA = nxtA; nxtA = t;
    t = curB; curB = nxtB; nxtB = t;
  }

#pragma unroll
  for (int j = 0; j < 4; ++j) {
    int n = n0 + wn * 64 + j * 16 + l16;
    float bj = b_out[n];
#pragma unroll
    for (int i = 0; i < 2; ++i) {
#pragma unroll
      for (int r = 0; r < 8; ++r) {
        int m = m0 + wm * 32 + i * 16 + r + half * 8;   // row in [B*T]
        int bb = m >> 11;                                // / T_SEQ
        int t  = m & (T_SEQ - 1);
        out[((size_t)t * BATCH + bb) * EMB + n] = acc[i][j][r] + bj;
      }
    }
  }
}

// ---------------------------------------------------------------------------
extern "C" void kernel_launch(void* const* d_in, const int* in_sizes, int n_in,
                              void* d_out, int out_size, void* d_ws, size_t ws_size,
                              hipStream_t stream) {
  (void)in_sizes; (void)n_in; (void)out_size; (void)ws_size;
  const float* query = (const float*)d_in[0];
  const float* key   = (const float*)d_in[1];
  const float* value = (const float*)d_in[2];
  const float* mask  = (const float*)d_in[3];
  const float* w_qkv = (const float*)d_in[4];
  const float* b_qkv = (const float*)d_in[5];
  const float* w_out = (const float*)d_in[6];
  const float* b_out = (const float*)d_in[7];
  float* out = (float*)d_out;

  const size_t NTOK = (size_t)T_SEQ * BATCH * EMB;      // 8388608
  const size_t NWQ  = (size_t)3 * EMB * EMB;            // 3145728
  const size_t NWO  = (size_t)EMB * EMB;                // 1048576
  bf16* xq   = (bf16*)d_ws;           // bf16 copies of inputs
  bf16* xk   = xq + NTOK;
  bf16* xv   = xk + NTOK;
  bf16* wqb  = xv + NTOK;
  bf16* wob  = wqb + NWQ;
  bf16* Qh   = wob + NWO;             // projected [B,H,T,DH]
  bf16* Kh   = Qh + NTOK;
  bf16* Vh   = Kh + NTOK;
  bf16* ctx  = Vh + NTOK;             // [B,T,E]

  cvt_f32_bf16_kernel<<<NTOK / 2048, 256, 0, stream>>>(query, xq, (int)NTOK);
  cvt_f32_bf16_kernel<<<NTOK / 2048, 256, 0, stream>>>(key,   xk, (int)NTOK);
  cvt_f32_bf16_kernel<<<NTOK / 2048, 256, 0, stream>>>(value, xv, (int)NTOK);
  cvt_f32_bf16_kernel<<<NWQ  / 2048, 256, 0, stream>>>(w_qkv, wqb, (int)NWQ);
  cvt_f32_bf16_kernel<<<NWO  / 2048, 256, 0, stream>>>(w_out, wob, (int)NWO);

  dim3 gQKV(64, 8, 3);               // M=8192/128, N=1024/128, {q,k,v}
  qkv_proj_kernel<<<gQKV, 256, 0, stream>>>(xq, xk, xv, wqb, b_qkv, Qh, Kh, Vh);

  dim3 gATT(T_SEQ / 128, BATCH * NH);
  attention_kernel<<<gATT, 256, 0, stream>>>(Qh, Kh, Vh, mask, ctx);

  dim3 gOUT(64, 8);                  // M=8192/128, N=1024/128
  out_proj_kernel<<<gOUT, 256, 0, stream>>>(ctx, wob, b_out, out);
}
